// DinoV2Attention_79903571574925
// MI455X (gfx1250) — matrix operand order repaired and checked by reading, not compiled, not run on hardware
//
#include <hip/hip_runtime.h>
#include <cstdint>
#include <cstddef>

// ---------------------------------------------------------------------------
// DinoV2 attention for MI455X (gfx1250, wave32, WMMA + async-to-LDS).
// All heavy math on v_wmma_f32_16x16x32_bf16 (f32 accumulate).
// Data movement: GLOBAL_LOAD_ASYNC_TO_LDS_B128 (ASYNCcnt) with double-
// buffered LDS tiles in the GEMMs; flash attention overlaps K-tile async
// DMA with the in-register V transpose.
// ---------------------------------------------------------------------------

#define HIDDEN   1024
#define HEADS    16
#define HEAD_DIM 64
#define BATCH    8
#define SEQ      1370
#define NTOK     (BATCH * SEQ)   // 10960

typedef __attribute__((ext_vector_type(16))) __bf16 v16bf;
typedef __attribute__((ext_vector_type(8)))  float  v8f;

union FragBF {            // 32 bytes: two 16B chunks <-> one WMMA A/B operand
    uint4  u[2];
    v16bf  v;
};

__device__ __forceinline__ v8f zero_v8f() {
    v8f z;
#pragma unroll
    for (int i = 0; i < 8; ++i) z[i] = 0.0f;
    return z;
}

__device__ __forceinline__ unsigned short f2bf(float f) {
    unsigned int x = __float_as_uint(f);
    x += 0x7FFFu + ((x >> 16) & 1u);          // round-to-nearest-even
    return (unsigned short)(x >> 16);
}

// --- gfx1250 async global->LDS (GVS mode: saddr base + 32-bit lane offset) --
__device__ __forceinline__ void async_ld_b128(unsigned lds_byte_off,
                                              const void* sbase,
                                              unsigned gbyte_off) {
    asm volatile("global_load_async_to_lds_b128 %0, %1, %2 offset:0"
                 :
                 : "v"(lds_byte_off), "v"(gbyte_off), "s"(sbase)
                 : "memory");
}
__device__ __forceinline__ void wait_async0() {
    asm volatile("s_wait_asynccnt 0x0" ::: "memory");
}
__device__ __forceinline__ void wait_async4() {   // allow 4 in flight (next tile)
    asm volatile("s_wait_asynccnt 0x4" ::: "memory");
}
__device__ __forceinline__ unsigned lds_off(const void* p) {
    return (unsigned)(size_t)p;   // low 32 bits of shared aperture = LDS offset
}

// ---------------------------------------------------------------------------
// f32 -> bf16 bulk convert (grid-stride)
// ---------------------------------------------------------------------------
__global__ void k_f32_to_bf16(const float* __restrict__ src,
                              unsigned short* __restrict__ dst, int n) {
    int i = blockIdx.x * blockDim.x + threadIdx.x;
    int stride = gridDim.x * blockDim.x;
    for (; i < n; i += stride) dst[i] = f2bf(src[i]);
}

// ---------------------------------------------------------------------------
// W[k][n] (f32, 1024x1024) -> WT[n][k] (bf16, N-major) via 32x32 LDS tiles
// ---------------------------------------------------------------------------
__global__ void k_transpose_bf16(const float* __restrict__ W,
                                 unsigned short* __restrict__ WT) {
    __shared__ float t[32][33];
    const int bx = blockIdx.x * 32;   // n
    const int by = blockIdx.y * 32;   // k
    const int tx = threadIdx.x;       // 0..31
    for (int j = threadIdx.y; j < 32; j += 8)
        t[j][tx] = W[(size_t)(by + j) * HIDDEN + bx + tx];
    __syncthreads();
    for (int j = threadIdx.y; j < 32; j += 8)
        WT[(size_t)(bx + j) * HIDDEN + by + tx] = f2bf(t[tx][j]);
}

// ---------------------------------------------------------------------------
// C[M,1024] = A[M,1024](bf16) * W (via WT[n][k] bf16) + bias
// Block tile 128x128, 8 waves, wave tile 64x32 (4x2 WMMA accumulators).
// Double-buffered LDS tiles filled by async-to-LDS DMA; the k-loop always
// has the next tile's 4 async ops/wave in flight while WMMAs run.
// ---------------------------------------------------------------------------
template <bool OUT_F32>
__global__ __launch_bounds__(256)
void k_gemm_bf16(const unsigned short* __restrict__ A,
                 const unsigned short* __restrict__ BT,
                 const float* __restrict__ bias,
                 void* __restrict__ Cout, int M) {
    __shared__ unsigned short As[2][128][40];   // 32 K + 8 pad, 2 buffers
    __shared__ unsigned short Bs[2][128][40];

    const int tid  = threadIdx.x;
    const int lane = tid & 31;
    const int wave = tid >> 5;
    const int wm   = (wave >> 2) * 64;  // wave M offset: 0 / 64
    const int wn   = (wave & 3) * 32;   // wave N offset: 0/32/64/96
    const int mblk = blockIdx.x * 128;
    const int nblk = blockIdx.y * 128;

    v8f acc[4][2];
#pragma unroll
    for (int i = 0; i < 4; ++i)
#pragma unroll
        for (int j = 0; j < 2; ++j) acc[i][j] = zero_v8f();

    // async-stage one 128x32 A tile + 128x32 B tile into buffer `buf`
    auto stage = [&](int buf, int k0) {
#pragma unroll
        for (int c = tid; c < 512; c += 256) {
            const int r = c >> 2, cc = c & 3;
            int m = mblk + r; if (m >= M) m = M - 1;        // edge clamp
            async_ld_b128(lds_off(&As[buf][r][cc * 8]), A,
                          (unsigned)(((size_t)m * HIDDEN + k0 + cc * 8) * 2));
            const int n = nblk + r;
            async_ld_b128(lds_off(&Bs[buf][r][cc * 8]), BT,
                          (unsigned)(((size_t)n * HIDDEN + k0 + cc * 8) * 2));
        }
    };

    stage(0, 0);                                            // prologue
    const int NK = HIDDEN / 32;
    for (int kk = 0; kk < NK; ++kk) {
        const int cur = kk & 1;
        if (kk + 1 < NK) { stage(cur ^ 1, (kk + 1) * 32); wait_async4(); }
        else             { wait_async0(); }
        __syncthreads();                                    // tile kk visible

        FragBF a[4], b[2];
        const int klo = (lane < 16) ? 0 : 8;   // A: K {0..7,16..23}/{8..15,24..31}
#pragma unroll
        for (int mt = 0; mt < 4; ++mt) {
            const int r = wm + mt * 16 + (lane & 15);
            a[mt].u[0] = *reinterpret_cast<const uint4*>(&As[cur][r][klo]);
            a[mt].u[1] = *reinterpret_cast<const uint4*>(&As[cur][r][klo + 16]);
        }
        const int kb = (lane < 16) ? 0 : 8;    // B: K {0..7,16..23}/{8..15,24..31}
#pragma unroll
        for (int nt = 0; nt < 2; ++nt) {
            const int r = wn + nt * 16 + (lane & 15);
            b[nt].u[0] = *reinterpret_cast<const uint4*>(&Bs[cur][r][kb]);
            b[nt].u[1] = *reinterpret_cast<const uint4*>(&Bs[cur][r][kb + 16]);
        }
#pragma unroll
        for (int mt = 0; mt < 4; ++mt)
#pragma unroll
            for (int nt = 0; nt < 2; ++nt)
                acc[mt][nt] = __builtin_amdgcn_wmma_f32_16x16x32_bf16(
                    false, a[mt].v, false, b[nt].v, (short)0, acc[mt][nt],
                    false, false);
        __syncthreads();   // all waves done reading buf[cur] before reuse
    }

    // C/D layout: VGPR i -> row i (+8 for lanes 16..31), col = lane&15
    const int rsel = (lane >> 4) * 8;
#pragma unroll
    for (int mt = 0; mt < 4; ++mt)
#pragma unroll
        for (int nt = 0; nt < 2; ++nt) {
            const int n  = nblk + wn + nt * 16 + (lane & 15);
            const float bv = bias[n];
#pragma unroll
            for (int i = 0; i < 8; ++i) {
                const int m = mblk + wm + mt * 16 + rsel + i;
                if (m < M) {
                    const float v = acc[mt][nt][i] + bv;
                    if (OUT_F32)
                        reinterpret_cast<float*>(Cout)[(size_t)m * HIDDEN + n] = v;
                    else
                        reinterpret_cast<unsigned short*>(Cout)[(size_t)m * HIDDEN + n] = f2bf(v);
                }
            }
        }
}

// ---------------------------------------------------------------------------
// Flash attention: one workgroup = one (b,h) x 128-query tile, 8 waves,
// each wave owns 16 query rows. K/V streamed in 64-key tiles through LDS.
// Q and K tiles arrive via async-to-LDS DMA; the V transpose (which needs
// VGPRs anyway) overlaps the K DMA.
// ---------------------------------------------------------------------------
__global__ __launch_bounds__(256)
void k_flash(const unsigned short* __restrict__ Qb,
             const unsigned short* __restrict__ Kb,
             const unsigned short* __restrict__ Vb,
             unsigned short* __restrict__ Ab) {
    __shared__ unsigned short Qs[128][72];     // [q][dh]
    __shared__ unsigned short Ks[64][72];      // [key][dh]
    __shared__ unsigned short Vt[64][72];      // [dh][key] (transposed)
    __shared__ unsigned short Ps[8][16][72];   // per-wave P staging [row][key]

    const int tid   = threadIdx.x;
    const int lane  = tid & 31;
    const int w     = tid >> 5;
    const int b     = blockIdx.x >> 4;
    const int h     = blockIdx.x & 15;
    const int qbase = blockIdx.y * 128;
    const float scale = 0.125f;                // 1/sqrt(64)

    const uint4* V4 = reinterpret_cast<const uint4*>(Vb);

    // async-stage Q tile [128 x 64]
#pragma unroll
    for (int c = tid; c < 1024; c += 256) {
        const int r = c >> 3, cc = c & 7;
        int s = qbase + r; if (s >= SEQ) s = SEQ - 1;
        async_ld_b128(lds_off(&Qs[r][cc * 8]), Qb,
                      (unsigned)(((size_t)(b * SEQ + s) * HIDDEN + h * 64 + cc * 8) * 2));
    }
    wait_async0();
    __syncthreads();

    // Q A-fragments are loop-invariant: load once
    FragBF aq[2];
    {
        const int r = w * 16 + (lane & 15);
#pragma unroll
        for (int kd = 0; kd < 2; ++kd) {
            const int klo = kd * 32 + ((lane < 16) ? 0 : 8);
            aq[kd].u[0] = *reinterpret_cast<const uint4*>(&Qs[r][klo]);
            aq[kd].u[1] = *reinterpret_cast<const uint4*>(&Qs[r][klo + 16]);
        }
    }

    v8f O[4];
#pragma unroll
    for (int ot = 0; ot < 4; ++ot) O[ot] = zero_v8f();
    float mrun[8], lrun[8];
#pragma unroll
    for (int i = 0; i < 8; ++i) { mrun[i] = -3.0e38f; lrun[i] = 0.0f; }

    const int nkv = (SEQ + 63) / 64;           // 22 key tiles
    for (int it = 0; it < nkv; ++it) {
        const int kvbase = it * 64;
        __syncthreads();
        // async K [64x64]; V loaded to VGPRs and scattered transposed while
        // the K DMA is in flight.
#pragma unroll
        for (int c = tid; c < 512; c += 256) {
            const int r = c >> 3, cc = c & 7;
            int s = kvbase + r; if (s >= SEQ) s = SEQ - 1;
            async_ld_b128(lds_off(&Ks[r][cc * 8]), Kb,
                          (unsigned)(((size_t)(b * SEQ + s) * HIDDEN + h * 64 + cc * 8) * 2));
        }
#pragma unroll
        for (int c = tid; c < 512; c += 256) {
            const int r = c >> 3, cc = c & 7;
            int s = kvbase + r; if (s >= SEQ) s = SEQ - 1;
            const uint4 vv =
                V4[(((size_t)(b * SEQ + s) * HIDDEN + h * 64) >> 3) + cc];
            const unsigned int vr[4] = {vv.x, vv.y, vv.z, vv.w};
#pragma unroll
            for (int j = 0; j < 4; ++j) {
                Vt[cc * 8 + 2 * j][r]     = (unsigned short)(vr[j] & 0xFFFFu);
                Vt[cc * 8 + 2 * j + 1][r] = (unsigned short)(vr[j] >> 16);
            }
        }
        wait_async0();
        __syncthreads();

        // scores: 4 tiles of 16 keys, K-dim = head_dim (2 wmma steps)
        float p[4][8];
#pragma unroll
        for (int t = 0; t < 4; ++t) {
            v8f sc = zero_v8f();
#pragma unroll
            for (int kd = 0; kd < 2; ++kd) {
                FragBF bk;
                const int r  = t * 16 + (lane & 15);
                const int cb = kd * 32 + ((lane < 16) ? 0 : 8);
                bk.u[0] = *reinterpret_cast<const uint4*>(&Ks[r][cb]);
                bk.u[1] = *reinterpret_cast<const uint4*>(&Ks[r][cb + 16]);
                sc = __builtin_amdgcn_wmma_f32_16x16x32_bf16(
                    false, aq[kd].v, false, bk.v, (short)0, sc, false, false);
            }
            const int col = kvbase + t * 16 + (lane & 15);
#pragma unroll
            for (int i = 0; i < 8; ++i)
                p[t][i] = (col < SEQ) ? sc[i] * scale : -1.0e30f;
        }

        // online softmax; rows of the C/D layout live in 16-lane halves,
        // so xor-shuffles 1..8 reduce exactly one row.
#pragma unroll
        for (int i = 0; i < 8; ++i) {
            float rm = fmaxf(fmaxf(p[0][i], p[1][i]), fmaxf(p[2][i], p[3][i]));
#pragma unroll
            for (int off = 1; off < 16; off <<= 1)
                rm = fmaxf(rm, __shfl_xor(rm, off, 32));
            const float mnew = fmaxf(mrun[i], rm);
            const float corr = __expf(mrun[i] - mnew);
            float rs = 0.0f;
#pragma unroll
            for (int t = 0; t < 4; ++t) {
                p[t][i] = __expf(p[t][i] - mnew);
                rs += p[t][i];
            }
#pragma unroll
            for (int off = 1; off < 16; off <<= 1)
                rs += __shfl_xor(rs, off, 32);
            lrun[i] = lrun[i] * corr + rs;
            mrun[i] = mnew;
#pragma unroll
            for (int ot = 0; ot < 4; ++ot) O[ot][i] *= corr;
        }

        // stage P (bf16) to re-layout C/D distribution -> A-fragment layout
        {
            const int prow = (lane >> 4) * 8;
#pragma unroll
            for (int t = 0; t < 4; ++t)
#pragma unroll
                for (int i = 0; i < 8; ++i)
                    Ps[w][prow + i][t * 16 + (lane & 15)] = f2bf(p[t][i]);
        }
        __syncthreads();

        // O += P[16x64] @ V[64x64]
#pragma unroll
        for (int kc = 0; kc < 2; ++kc) {
            FragBF ap;
            const int r  = lane & 15;
            const int cb = kc * 32 + ((lane < 16) ? 0 : 8);
            ap.u[0] = *reinterpret_cast<const uint4*>(&Ps[w][r][cb]);
            ap.u[1] = *reinterpret_cast<const uint4*>(&Ps[w][r][cb + 16]);
#pragma unroll
            for (int ot = 0; ot < 4; ++ot) {
                FragBF bv;
                const int dh = ot * 16 + (lane & 15);
                const int vb = kc * 32 + ((lane < 16) ? 0 : 8);
                bv.u[0] = *reinterpret_cast<const uint4*>(&Vt[dh][vb]);
                bv.u[1] = *reinterpret_cast<const uint4*>(&Vt[dh][vb + 16]);
                O[ot] = __builtin_amdgcn_wmma_f32_16x16x32_bf16(
                    false, ap.v, false, bv.v, (short)0, O[ot], false, false);
            }
        }
    }

    // normalize and write attention output (bf16, [N_tok,1024] layout)
    const int prow = (lane >> 4) * 8;
#pragma unroll
    for (int i = 0; i < 8; ++i) {
        const float inv = 1.0f / lrun[i];
        const int s = qbase + w * 16 + prow + i;
        if (s < SEQ) {
            const size_t rowoff = (size_t)(b * SEQ + s) * HIDDEN + h * 64;
#pragma unroll
            for (int ot = 0; ot < 4; ++ot)
                Ab[rowoff + ot * 16 + (lane & 15)] = f2bf(O[ot][i] * inv);
        }
    }
}

// ---------------------------------------------------------------------------
// Launch
// ---------------------------------------------------------------------------
extern "C" void kernel_launch(void* const* d_in, const int* in_sizes, int n_in,
                              void* d_out, int out_size, void* d_ws,
                              size_t ws_size, hipStream_t stream) {
    (void)in_sizes; (void)n_in; (void)out_size; (void)ws_size;

    const float* x  = (const float*)d_in[0];
    const float* Wq = (const float*)d_in[1];
    const float* bq = (const float*)d_in[2];
    const float* Wk = (const float*)d_in[3];
    const float* bk = (const float*)d_in[4];
    const float* Wv = (const float*)d_in[5];
    const float* bv = (const float*)d_in[6];
    const float* Wo = (const float*)d_in[7];
    const float* bo = (const float*)d_in[8];
    float* out = (float*)d_out;

    // workspace layout (bf16 everywhere): ~120 MB total
    char* ws = (char*)d_ws;
    const size_t szTok = (size_t)NTOK * HIDDEN * sizeof(unsigned short);
    const size_t szW   = (size_t)HIDDEN * HIDDEN * sizeof(unsigned short);
    unsigned short* xb  = (unsigned short*)(ws);
    unsigned short* WqT = (unsigned short*)(ws + szTok);
    unsigned short* WkT = (unsigned short*)(ws + szTok + szW);
    unsigned short* WvT = (unsigned short*)(ws + szTok + 2 * szW);
    unsigned short* WoT = (unsigned short*)(ws + szTok + 3 * szW);
    unsigned short* Qb  = (unsigned short*)(ws + szTok + 4 * szW);
    unsigned short* Kb  = (unsigned short*)(ws + 2 * szTok + 4 * szW);
    unsigned short* Vb  = (unsigned short*)(ws + 3 * szTok + 4 * szW);
    unsigned short* Ab  = (unsigned short*)(ws + 4 * szTok + 4 * szW);

    // 1) x -> bf16
    k_f32_to_bf16<<<2048, 256, 0, stream>>>(x, xb, NTOK * HIDDEN);

    // 2) weights -> transposed bf16 (N-major)
    dim3 tg(HIDDEN / 32, HIDDEN / 32), tb(32, 8);
    k_transpose_bf16<<<tg, tb, 0, stream>>>(Wq, WqT);
    k_transpose_bf16<<<tg, tb, 0, stream>>>(Wk, WkT);
    k_transpose_bf16<<<tg, tb, 0, stream>>>(Wv, WvT);
    k_transpose_bf16<<<tg, tb, 0, stream>>>(Wo, WoT);

    // 3) Q/K/V projections (bf16 out)
    dim3 gg((NTOK + 127) / 128, HIDDEN / 128);
    k_gemm_bf16<false><<<gg, 256, 0, stream>>>(xb, WqT, bq, Qb, NTOK);
    k_gemm_bf16<false><<<gg, 256, 0, stream>>>(xb, WkT, bk, Kb, NTOK);
    k_gemm_bf16<false><<<gg, 256, 0, stream>>>(xb, WvT, bv, Vb, NTOK);

    // 4) flash attention
    dim3 fg(BATCH * HEADS, (SEQ + 127) / 128);
    k_flash<<<fg, 256, 0, stream>>>(Qb, Kb, Vb, Ab);

    // 5) output projection (f32 out + bias)
    k_gemm_bf16<true><<<gg, 256, 0, stream>>>(Ab, WoT, bo, out, NTOK);
}
